// EnhancedNNet_9448928051214
// MI455X (gfx1250) — matrix operand order repaired and checked by reading, not compiled, not run on hardware
//
#include <hip/hip_runtime.h>

typedef __attribute__((ext_vector_type(16))) _Float16 v16h;
typedef __attribute__((ext_vector_type(8)))  _Float16 v8h;
typedef __attribute__((ext_vector_type(8)))  float    v8f;
typedef __attribute__((ext_vector_type(4)))  int      v4i;

typedef __attribute__((address_space(1))) v4i GV4;   // global int4
typedef __attribute__((address_space(3))) v4i LV4;   // LDS int4

#define NB  32
#define NN  1024
#define DD  64
#define AA  256
#define BPAD 1032   // LDS row stride (halves): 1024 + 8 -> conflict-free b128 reads

// ---------------------------------------------------------------------------
// Stage 1: batch-independent MLP + first graph projection.
//   h  = relu(relu(s@W1+b1)@W2+b2)            (1024 x 64)
//   support1 = h@Wg1 + bg1                    (1024 x 64)  [no relu]
// Stored TRANSPOSED as f16: sup1T[d][m] — contiguous-K for the WMMA B-frag.
// ---------------------------------------------------------------------------
__global__ __launch_bounds__(128) void mlp_kernel(
    const float* __restrict__ ns,
    const float* __restrict__ W1, const float* __restrict__ b1,
    const float* __restrict__ W2, const float* __restrict__ b2,
    const float* __restrict__ Wg1, const float* __restrict__ bg1,
    _Float16* __restrict__ sup1T)
{
    __shared__ float srow[256];
    __shared__ float h1[128];
    __shared__ float h2[64];
    const int t   = threadIdx.x;
    const int row = blockIdx.x;

    srow[t]       = ns[row * 256 + t];
    srow[t + 128] = ns[row * 256 + t + 128];
    __syncthreads();

    float acc = b1[t];
#pragma unroll 8
    for (int k = 0; k < 256; k++) acc = fmaf(srow[k], W1[k * 128 + t], acc);
    h1[t] = fmaxf(acc, 0.f);
    __syncthreads();

    if (t < 64) {
        float a2 = b2[t];
#pragma unroll 8
        for (int k = 0; k < 128; k++) a2 = fmaf(h1[k], W2[k * 64 + t], a2);
        h2[t] = fmaxf(a2, 0.f);
    }
    __syncthreads();

    if (t < 64) {
        float a3 = bg1[t];
#pragma unroll 8
        for (int k = 0; k < 64; k++) a3 = fmaf(h2[k], Wg1[k * 64 + t], a3);
        sup1T[t * NN + row] = (_Float16)a3;
    }
}

// ---------------------------------------------------------------------------
// Stage 2: e1T[b][d][m] = relu(adj[b] @ support1)^T,  32x(1024x1024x64).
//  - adjacency (134MB f32) streamed from HBM once, cvt to f16 in regs
//  - support1^T (128KB f16) staged into LDS ONCE per workgroup via the
//    gfx1250 async global->LDS path (ASYNCcnt), then read with ds_load_b128
//  - one wave = 16x64 strip: 1 A-frag feeds 4 v_wmma per k-step
//  - C written transposed: each accumulator's 8 rows are contiguous -> b128s
// ---------------------------------------------------------------------------
__global__ __launch_bounds__(128) void gcn1_wmma_kernel(
    const float* __restrict__ adj,
    const _Float16* __restrict__ supT,
    float* __restrict__ e1T)
{
    extern __shared__ _Float16 smem[];   // 64 * BPAD halves = 129 KB

    const int t       = threadIdx.x;
    const int lane    = t & 31;
    const int wave    = t >> 5;
    const int b       = blockIdx.y;
    const int rowbase = (blockIdx.x * 4 + wave) * 16;
    const int m       = lane & 15;   // row (A) / column (B/C) within tile
    const int hi      = lane >> 4;   // lane-group select

    // ---- stage support1^T into LDS (column col: global stride 1024 -> padded 1032)
    {
        const int col  = t >> 1;          // 0..63
        const int half = t & 1;           // 0..1 (512 halves each)
        const _Float16* g = supT + col * NN   + half * 512;
        _Float16*       l = smem + col * BPAD + half * 512;
#if __has_builtin(__builtin_amdgcn_global_load_async_to_lds_b128)
#pragma unroll
        for (int i = 0; i < 64; i++)
            __builtin_amdgcn_global_load_async_to_lds_b128(
                (GV4*)(g + i * 8), (LV4*)(l + i * 8), 0, 0);
#if __has_builtin(__builtin_amdgcn_s_wait_asynccnt)
        __builtin_amdgcn_s_wait_asynccnt(0);
#else
        asm volatile("s_wait_asynccnt 0" ::: "memory");
#endif
#else
        for (int i = 0; i < 512; i++) l[i] = g[i];
#endif
    }
    __syncthreads();

    // A fragment source: adj[b, rowbase+m, kb + hi*16 .. +15] (two 8-float runs)
    const float* ap = adj + ((size_t)b * NN + rowbase + m) * NN + hi * 16;
    // B fragment sources in LDS: smem[col][kb + hi*16 .. +15]
    const _Float16* bq0 = smem + ( 0 + m) * BPAD + hi * 16;
    const _Float16* bq1 = smem + (16 + m) * BPAD + hi * 16;
    const _Float16* bq2 = smem + (32 + m) * BPAD + hi * 16;
    const _Float16* bq3 = smem + (48 + m) * BPAD + hi * 16;

    v8f c0 = {}; v8f c1 = {}; v8f c2 = {}; v8f c3 = {};

    for (int kb = 0; kb < NN; kb += 32) {
        v8f alo = *(const v8f*)(ap + kb);
        v8f ahi = *(const v8f*)(ap + kb + 8);
        v16h a;
#pragma unroll
        for (int i = 0; i < 8; i++) {
            a[i]     = (_Float16)alo[i];
            a[8 + i] = (_Float16)ahi[i];
        }
        v8h b0l = *(const v8h*)(bq0 + kb), b0h = *(const v8h*)(bq0 + kb + 8);
        v8h b1l = *(const v8h*)(bq1 + kb), b1h = *(const v8h*)(bq1 + kb + 8);
        v8h b2l = *(const v8h*)(bq2 + kb), b2h = *(const v8h*)(bq2 + kb + 8);
        v8h b3l = *(const v8h*)(bq3 + kb), b3h = *(const v8h*)(bq3 + kb + 8);
        v16h bf0 = __builtin_shufflevector(b0l, b0h, 0,1,2,3,4,5,6,7,8,9,10,11,12,13,14,15);
        v16h bf1 = __builtin_shufflevector(b1l, b1h, 0,1,2,3,4,5,6,7,8,9,10,11,12,13,14,15);
        v16h bf2 = __builtin_shufflevector(b2l, b2h, 0,1,2,3,4,5,6,7,8,9,10,11,12,13,14,15);
        v16h bf3 = __builtin_shufflevector(b3l, b3h, 0,1,2,3,4,5,6,7,8,9,10,11,12,13,14,15);
        c0 = __builtin_amdgcn_wmma_f32_16x16x32_f16(false, a, false, bf0, (short)0, c0, false, false);
        c1 = __builtin_amdgcn_wmma_f32_16x16x32_f16(false, a, false, bf1, (short)0, c1, false, false);
        c2 = __builtin_amdgcn_wmma_f32_16x16x32_f16(false, a, false, bf2, (short)0, c2, false, false);
        c3 = __builtin_amdgcn_wmma_f32_16x16x32_f16(false, a, false, bf3, (short)0, c3, false, false);
    }

    // C/D: element i -> row rowbase + hi*8 + i, col nt*16 + m.
    // e1T[b][col][row]: 8 rows contiguous per accumulator -> vector stores.
    const size_t rb = rowbase + hi * 8;
    float* op0 = e1T + ((size_t)b * DD +  0 + m) * NN + rb;
    float* op1 = e1T + ((size_t)b * DD + 16 + m) * NN + rb;
    float* op2 = e1T + ((size_t)b * DD + 32 + m) * NN + rb;
    float* op3 = e1T + ((size_t)b * DD + 48 + m) * NN + rb;
    v8f r0, r1, r2, r3;
#pragma unroll
    for (int i = 0; i < 8; i++) {
        r0[i] = fmaxf(c0[i], 0.f);
        r1[i] = fmaxf(c1[i], 0.f);
        r2[i] = fmaxf(c2[i], 0.f);
        r3[i] = fmaxf(c3[i], 0.f);
    }
    *(v8f*)op0 = r0;
    *(v8f*)op1 = r1;
    *(v8f*)op2 = r2;
    *(v8f*)op3 = r3;
}

// ---------------------------------------------------------------------------
// Stage 3: layer 2 collapses through the e[:,0,:] slice:
//   r[b][d] = adj[b,0,:] . e1T[b][d][:]     (coalesced along m)
//   asum[b] = sum(adj[b,0,:])
// 8 waves/block; wave w owns d = 8w..8w+7; wave32 shfl reduction.
// ---------------------------------------------------------------------------
__global__ __launch_bounds__(256) void reduce_kernel(
    const float* __restrict__ adj,
    const float* __restrict__ e1T,
    float* __restrict__ r,
    float* __restrict__ asum)
{
    __shared__ float reda[256];
    const int t    = threadIdx.x;
    const int b    = blockIdx.x;
    const int lane = t & 31;
    const int wave = t >> 5;

    const float* arow = adj + (size_t)b * NN * NN;  // adj[b, 0, :]

#pragma unroll
    for (int i = 0; i < 8; i++) {
        const int d = wave * 8 + i;
        const float* ed = e1T + ((size_t)b * DD + d) * NN;
        float acc = 0.f;
        for (int mm = lane; mm < NN; mm += 32)
            acc = fmaf(arow[mm], ed[mm], acc);
#pragma unroll
        for (int off = 16; off > 0; off >>= 1)
            acc += __shfl_down(acc, off, 32);
        if (lane == 0) r[b * 64 + d] = acc;
    }

    float sa = 0.f;
    for (int mm = t; mm < NN; mm += 256) sa += arow[mm];
    reda[t] = sa;
    for (int s = 128; s > 0; s >>= 1) {
        __syncthreads();
        if (t < s) reda[t] += reda[t + s];
    }
    __syncthreads();
    if (t == 0) asum[b] = reda[0];
}

// ---------------------------------------------------------------------------
// Stage 4: head.
//   cur = relu(r@Wg2 + asum*bg2); pi = softmax(cur@Wp+bp); v = tanh(cur@Wv+bv)
// Output: pi (32x256) then v (32) flat.
// ---------------------------------------------------------------------------
__global__ __launch_bounds__(256) void head_kernel(
    const float* __restrict__ r, const float* __restrict__ asum,
    const float* __restrict__ Wg2, const float* __restrict__ bg2,
    const float* __restrict__ Wp,  const float* __restrict__ bp,
    const float* __restrict__ Wv,  const float* __restrict__ bv,
    float* __restrict__ out)
{
    __shared__ float cur[64];
    __shared__ float red[256];
    const int t = threadIdx.x;
    const int b = blockIdx.x;

    if (t < 64) {
        float acc = asum[b] * bg2[t];
#pragma unroll 8
        for (int k = 0; k < 64; k++) acc = fmaf(r[b * 64 + k], Wg2[k * 64 + t], acc);
        cur[t] = fmaxf(acc, 0.f);
    }
    __syncthreads();

    float logit = bp[t];
#pragma unroll 8
    for (int k = 0; k < 64; k++) logit = fmaf(cur[k], Wp[k * 256 + t], logit);

    red[t] = logit;
    __syncthreads();
    for (int s = 128; s > 0; s >>= 1) {
        if (t < s) red[t] = fmaxf(red[t], red[t + s]);
        __syncthreads();
    }
    float mx = red[0];
    __syncthreads();
    float ex = __expf(logit - mx);
    red[t] = ex;
    __syncthreads();
    for (int s = 128; s > 0; s >>= 1) {
        if (t < s) red[t] += red[t + s];
        __syncthreads();
    }
    out[b * AA + t] = ex / red[0];

    if (t == 0) {
        float vv = bv[0];
        for (int k = 0; k < 64; k++) vv = fmaf(cur[k], Wv[k], vv);
        out[NB * AA + b] = tanhf(vv);
    }
}

// ---------------------------------------------------------------------------
extern "C" void kernel_launch(void* const* d_in, const int* in_sizes, int n_in,
                              void* d_out, int out_size, void* d_ws, size_t ws_size,
                              hipStream_t stream) {
    // 0:x 1:neighbor_states 2:adjacency 3:W1 4:b1 5:W2 6:b2
    // 7:Wg1 8:bg1 9:Wg2 10:bg2 11:Wp 12:bp 13:Wv 14:bv
    const float* ns  = (const float*)d_in[1];
    const float* adj = (const float*)d_in[2];
    const float* W1  = (const float*)d_in[3];
    const float* b1  = (const float*)d_in[4];
    const float* W2  = (const float*)d_in[5];
    const float* b2  = (const float*)d_in[6];
    const float* Wg1 = (const float*)d_in[7];
    const float* bg1 = (const float*)d_in[8];
    const float* Wg2 = (const float*)d_in[9];
    const float* bg2 = (const float*)d_in[10];
    const float* Wp  = (const float*)d_in[11];
    const float* bp  = (const float*)d_in[12];
    const float* Wv  = (const float*)d_in[13];
    const float* bv  = (const float*)d_in[14];

    char* ws = (char*)d_ws;
    _Float16* sup1T = (_Float16*)ws;                               // 128 KB
    float*    e1T   = (float*)(ws + (64 * 1024 * 2));              // 8 MB
    float*    rvec  = (float*)(ws + (64 * 1024 * 2) + (32u * 1024 * 64 * 4));
    float*    asum  = (float*)(ws + (64 * 1024 * 2) + (32u * 1024 * 64 * 4) + 8192);

    mlp_kernel<<<NN, 128, 0, stream>>>(ns, W1, b1, W2, b2, Wg1, bg1, sup1T);
    gcn1_wmma_kernel<<<dim3(NN / 64, NB), 128, 64 * BPAD * sizeof(_Float16), stream>>>(
        adj, sup1T, e1T);
    reduce_kernel<<<NB, 256, 0, stream>>>(adj, e1T, rvec, asum);
    head_kernel<<<NB, 256, 0, stream>>>(rvec, asum, Wg2, bg2, Wp, bp, Wv, bv,
                                        (float*)d_out);
}
